// GraphRNN_36404142801721
// MI455X (gfx1250) — matrix operand order repaired and checked by reading, not compiled, not run on hardware
//
#include <hip/hip_runtime.h>

#define N_NODES 65536
#define T_STEPS 64
#define H 256
#define ROWS 32           // nodes per workgroup
#define SH 264            // padded bf16 row stride for h state (bank rotation, 16B aligned)

typedef __attribute__((ext_vector_type(16))) __bf16 v16bf;
typedef __attribute__((ext_vector_type(8)))  __bf16 v8bf;
typedef __attribute__((ext_vector_type(8)))  float  v8f;
typedef __attribute__((ext_vector_type(4)))  float  v4f;

__device__ __forceinline__ float fsigmoid(float x) {
  float e = __builtin_amdgcn_exp2f(x * -1.44269504088896340736f);
  return __builtin_amdgcn_rcpf(1.0f + e);
}
__device__ __forceinline__ float ftanh(float x) {
  float e = __builtin_amdgcn_exp2f(x * -2.88539008177792681472f);
  return fmaf(2.0f, __builtin_amdgcn_rcpf(1.0f + e), -1.0f);
}

// GRU recurrence on WMMA: per step, gh = h @ W_hh^T via v_wmma_f32_16x16x32_bf16.
// W_hh^T B-fragments are register-resident per wave (each wave owns 32 hidden
// columns => 6 16-col tiles x 8 K-tiles = 48 v16bf = 384 VGPRs), loaded once.
__global__ __launch_bounds__(256, 1)
void graphrnn_edge_gru(const float* __restrict__ adj,
                       const float* __restrict__ Wih,
                       const float* __restrict__ Whh,
                       const float* __restrict__ bih,
                       const float* __restrict__ bhh,
                       const float* __restrict__ Wout,
                       const float* __restrict__ bout,
                       float* __restrict__ out)
{
  __shared__ __align__(16) __bf16 h16[2][ROWS * SH];   // ping-pong bf16 hidden state
  __shared__ __align__(16) float  adjL[ROWS * T_STEPS];
  __shared__ __align__(16) float  woutL[H];
  __shared__ __align__(16) float  outL[ROWS * T_STEPS];

  const int tid  = threadIdx.x;
  const int w    = tid >> 5;        // wave id 0..7: owns hidden cols [w*32, w*32+32)
  const int lane = tid & 31;
  const int l15  = lane & 15;
  const int half = lane >> 4;
  const long nodeBase = (long)blockIdx.x * ROWS;

  // ---- stage adjacency tile + W_out, zero h0 ----
  {
    const float* src = adj + nodeBase * T_STEPS;
    for (int i = tid; i < ROWS * T_STEPS / 4; i += 256)
      ((v4f*)adjL)[i] = ((const v4f*)src)[i];
    if (tid < H) woutL[tid] = Wout[tid];
    for (int i = tid; i < ROWS * SH; i += 256) h16[0][i] = (__bf16)0.0f;
  }

  // ---- per-lane gate constants + register-resident W_hh^T fragments ----
  // B frag as loaded here (contiguous k, same order as A): lane n%16 = output column, 16 consecutive K
  // per lane starting at kt*32 + (lane/16)*16  => contiguous run of W_hh row c.
  float wihv[3][2], bsum[2][2], bihn[2], bhhn[2];
  v16bf Bf[6][8];
  #pragma unroll
  for (int g = 0; g < 3; ++g) {
    #pragma unroll
    for (int p = 0; p < 2; ++p) {
      const int c = g * H + w * 32 + p * 16 + l15;     // column in [0,768)
      wihv[g][p] = Wih[c];
      if (g < 2) { bsum[g][p] = bih[c] + bhh[c]; }
      else       { bihn[p] = bih[c]; bhhn[p] = bhh[c]; }
      #pragma unroll
      for (int kt = 0; kt < 8; ++kt) {
        const float* wr = Whh + (long)c * H + kt * 32 + half * 16;
        v16bf b;
        #pragma unroll
        for (int i = 0; i < 16; ++i) b[i] = (__bf16)wr[i];
        Bf[g * 2 + p][kt] = b;
      }
    }
  }
  const float bo = bout[0];
  __syncthreads();

  for (int t = 0; t < T_STEPS; ++t) {
    const __bf16* hcur = h16[t & 1];
    __bf16*       hnxt = h16[(t + 1) & 1];

    #pragma unroll
    for (int mt = 0; mt < 2; ++mt) {
      const int m0 = mt * 16;
      v8f acc[6];
      #pragma unroll
      for (int j = 0; j < 6; ++j) acc[j] = (v8f){0.f,0.f,0.f,0.f,0.f,0.f,0.f,0.f};

      #pragma unroll
      for (int kh = 0; kh < 2; ++kh) {
        // A frag (16-bit 16x32): lane row = m0+l15; v0-3 <- K kt*32+half*16,
        // v4-7 <- K kt*32+half*16+8. Two ds_load_b128 per fragment.
        v16bf A[4];
        #pragma unroll
        for (int i = 0; i < 4; ++i) {
          const int kt = kh * 4 + i;
          const __bf16* rp = hcur + (m0 + l15) * SH + kt * 32 + half * 16;
          v8bf lo = *(const v8bf*)rp;
          v8bf hi = *(const v8bf*)(rp + 8);
          v16bf a;
          #pragma unroll
          for (int e = 0; e < 8; ++e) { a[e] = lo[e]; a[8 + e] = hi[e]; }
          A[i] = a;
        }
        #pragma unroll
        for (int j = 0; j < 6; ++j)
          #pragma unroll
          for (int i = 0; i < 4; ++i)
            acc[j] = __builtin_amdgcn_wmma_f32_16x16x32_bf16(
                false, A[i], false, Bf[j][kh * 4 + i], (short)0, acc[j],
                false, false);
      }

      // gate math on f32 accumulators; C/D layout: lane col = l15, row = v+8*half
      #pragma unroll
      for (int p = 0; p < 2; ++p) {
        const int hc = w * 32 + p * 16 + l15;
        #pragma unroll
        for (int v = 0; v < 8; ++v) {
          const int row = m0 + v + half * 8;
          const float x  = adjL[row * T_STEPS + t];
          const float rr = fsigmoid(acc[0 + p][v] + fmaf(x, wihv[0][p], bsum[0][p]));
          const float zz = fsigmoid(acc[2 + p][v] + fmaf(x, wihv[1][p], bsum[1][p]));
          const float nn = ftanh(fmaf(x, wihv[2][p], bihn[p]) +
                                 rr * (acc[4 + p][v] + bhhn[p]));
          const float ho = (float)hcur[row * SH + hc];
          const float hv = fmaf(zz, ho - nn, nn);      // (1-z)*n + z*h
          hnxt[row * SH + hc] = (__bf16)hv;
        }
      }
    }
    __syncthreads();   // all h_{t+1} written; all reads of h_t done

    // out[:, t] = sigmoid(h_{t+1} . W_out + b_out), one lane per node row
    if (tid < ROWS) {
      const __bf16* hr = hnxt + tid * SH;
      float s = bo;
      #pragma unroll 4
      for (int c = 0; c < H; c += 8) {
        v8bf hv = *(const v8bf*)(hr + c);
        #pragma unroll
        for (int e = 0; e < 8; ++e) s = fmaf((float)hv[e], woutL[c + e], s);
      }
      outL[tid * T_STEPS + t] = fsigmoid(s);
    }
    __syncthreads();   // outL done; safe to overwrite old buffer next step
  }

  // coalesced writeback of [ROWS x 64] outputs
  {
    float* dst = out + nodeBase * T_STEPS;
    for (int i = tid; i < ROWS * T_STEPS / 4; i += 256)
      ((v4f*)dst)[i] = ((const v4f*)outL)[i];
  }
}

extern "C" void kernel_launch(void* const* d_in, const int* in_sizes, int n_in,
                              void* d_out, int out_size, void* d_ws, size_t ws_size,
                              hipStream_t stream) {
  const float* adj  = (const float*)d_in[0];  // [65536, 64]
  const float* Wih  = (const float*)d_in[1];  // [768, 1]
  const float* Whh  = (const float*)d_in[2];  // [768, 256]
  const float* bihp = (const float*)d_in[3];  // [768]
  const float* bhhp = (const float*)d_in[4];  // [768]
  const float* Wout = (const float*)d_in[5];  // [1, 256]
  const float* bout = (const float*)d_in[6];  // [1]
  float* outp = (float*)d_out;                // [65536, 64]

  dim3 grid(N_NODES / ROWS), block(256);
  graphrnn_edge_gru<<<grid, block, 0, stream>>>(adj, Wih, Whh, bihp, bhhp,
                                                Wout, bout, outp);
}
